// GNN_SAG_39694087750183
// MI455X (gfx1250) — matrix operand order repaired and checked by reading, not compiled, not run on hardware
//
#include <hip/hip_runtime.h>
#include <hip/hip_bf16.h>

typedef __attribute__((ext_vector_type(16))) _Float16 v16h;
typedef __attribute__((ext_vector_type(8)))  float    v8f;

#define N_NODES   100000
#define N_EDGES   800000
#define N_FEAT    128
#define NHID      192
#define HEADS     3
#define HDIM      64
#define N_GRAPHS  64
#define BN_EPS    1e-5f
#define NEG_SLOPE 0.2f
#define NTILES    12      /* 192/16 */
#define MTILES    6250    /* 100000/16 */

static __device__ __forceinline__ void atomicMaxF32(float* addr, float v) {
  if (v >= 0.0f) atomicMax((int*)addr, __float_as_int(v));
  else           atomicMin((unsigned int*)addr, __float_as_uint(v));
}

__global__ void fill_f32(float* __restrict__ p, float val, int n) {
  int i = blockIdx.x * blockDim.x + threadIdx.x;
  if (i < n) p[i] = val;
}

__global__ void count_nodes(const int* __restrict__ batch, float* __restrict__ cnt) {
  int i = blockIdx.x * blockDim.x + threadIdx.x;
  if (i < N_NODES) atomicAdd(&cnt[batch[i]], 1.0f);
}

// Repack W [K,192] (f32, row-major) into per-wave WMMA B fragments (f16).
// B fragment order used here (16-bit, 32x16, contiguous k): lane holds col N = lane%16; lanes 0-15: K=0..15,
// lanes 16-31: K=16..31; VGPR v holds K = koff+2v, koff+2v+1.
__global__ void repack_w(const float* __restrict__ W, _Float16* __restrict__ wpk, int K) {
  int ksteps = K >> 5;
  int total = NTILES * ksteps * 32 * 16;
  int idx = blockIdx.x * blockDim.x + threadIdx.x;
  if (idx >= total) return;
  int e    = idx & 15;
  int lane = (idx >> 4) & 31;
  int s    = (idx >> 9) % ksteps;
  int n    = idx / (512 * ksteps);
  int v = e >> 1, p = e & 1;
  int kk  = s * 32 + ((lane >> 4) << 4) + 2 * v + p;
  int col = n * 16 + (lane & 15);
  wpk[idx] = (_Float16)W[kk * NHID + col];
}

// C[100000,192] = A[100000,K] @ W[K,192] via v_wmma_f32_16x16x32_f16.
// One block = 8 waves sharing one 6-N-tile group; B fragments for the whole
// group are staged once into LDS via gfx1250 async copies, then each wave's
// k-loop feeds its WMMAs from ds_load_b128 (no per-k-step global traffic).
__global__ __launch_bounds__(256) void gemm_wmma(const float* __restrict__ A,
                                                 const _Float16* __restrict__ wpk,
                                                 float* __restrict__ C, int K) {
  // 6 ntiles x (K/32 <= 6) ksteps x 32 lanes, one v16h (32B) fragment each
  __shared__ v16h sB[6 * 6 * 32];
  int ksteps = K >> 5;
  int nhalf  = blockIdx.x & 1;
  int mblock = blockIdx.x >> 1;
  int wave   = threadIdx.x >> 5;
  int lane   = threadIdx.x & 31;
  int nbase  = nhalf * 6;

  // ---- async-stage this block's B fragments (global -> LDS, 16B per lane-op)
  {
    const _Float16* src = wpk + (size_t)nbase * ksteps * 512;
    int chunks = (6 * ksteps * 512 * 2) / 16;           // 16-byte chunks
    unsigned lds0 = (unsigned)(size_t)(&sB[0]);
    for (int i = threadIdx.x; i < chunks; i += 256) {
      unsigned lds = lds0 + i * 16;
      unsigned long long ga = (unsigned long long)(size_t)src + (unsigned long long)i * 16;
      asm volatile("global_load_async_to_lds_b128 %0, %1, off"
                   :: "v"(lds), "v"(ga) : "memory");
    }
    asm volatile("s_wait_asynccnt 0" ::: "memory");
  }
  __syncthreads();

  int mtile = mblock * 8 + wave;
  if (mtile >= MTILES) return;                 // wave-uniform exit after barrier
  int row  = mtile * 16 + (lane & 15);
  int kgrp = (lane >> 4) * 16;                 // A-matrix half-lane K offset

  v8f acc[6];
#pragma unroll
  for (int j = 0; j < 6; ++j) acc[j] = (v8f){0.f,0.f,0.f,0.f,0.f,0.f,0.f,0.f};

  for (int s = 0; s < ksteps; ++s) {
    const float* arow = A + (long)row * K + s * 32;
    v16h af;
#pragma unroll
    for (int v = 0; v < 8; ++v) {
      int kk = kgrp + 2 * v;
      float2 t = *(const float2*)(arow + kk);
      af[2 * v]     = (_Float16)t.x;
      af[2 * v + 1] = (_Float16)t.y;
    }
#pragma unroll
    for (int j = 0; j < 6; ++j) {
      v16h bf = sB[(j * ksteps + s) * 32 + lane];
      acc[j] = __builtin_amdgcn_wmma_f32_16x16x32_f16(
          false, af, false, bf, (short)0, acc[j], false, false);
    }
  }

  int crow0 = mtile * 16 + ((lane >> 4) << 3);
  int col0  = lane & 15;
#pragma unroll
  for (int j = 0; j < 6; ++j) {
#pragma unroll
    for (int r = 0; r < 8; ++r) {
      C[(long)(crow0 + r) * NHID + (nbase + j) * 16 + col0] = acc[j][r];
    }
  }
}

// al_s[n,h] = <xp[n,h,:], a_src[h,:]>; al_d analogous.
__global__ void attn_coeff(const float* __restrict__ xp, const float* __restrict__ asrc,
                           const float* __restrict__ adst, float* __restrict__ als,
                           float* __restrict__ ald) {
  int idx = blockIdx.x * blockDim.x + threadIdx.x;
  if (idx >= N_NODES * HEADS) return;
  int n = idx / HEADS, h = idx % HEADS;
  const float* row = xp + (long)n * NHID + h * HDIM;
  const float* as_ = asrc + h * HDIM;
  const float* ad_ = adst + h * HDIM;
  float s = 0.f, d = 0.f;
#pragma unroll 4
  for (int i = 0; i < HDIM; ++i) { float v = row[i]; s += v * as_[i]; d += v * ad_[i]; }
  als[idx] = s; ald[idx] = d;
}

// e = leaky_relu(al_s[src] + al_d[dst]); segment max over dst via atomic max.
__global__ void edge_logit_max(const int* __restrict__ ei, const float* __restrict__ als,
                               const float* __restrict__ ald, float* __restrict__ eex,
                               float* __restrict__ m) {
  int idx = blockIdx.x * blockDim.x + threadIdx.x;
  if (idx >= N_EDGES * HEADS) return;
  int e = idx / HEADS, h = idx % HEADS;
  int s = ei[e], d = ei[N_EDGES + e];
  float v = als[s * HEADS + h] + ald[d * HEADS + h];
  v = (v > 0.f) ? v : NEG_SLOPE * v;
  eex[idx] = v;
  atomicMaxF32(&m[d * HEADS + h], v);
}

// ex = exp(e - m[dst]); segment sum into den.
__global__ void edge_exp_sum(const int* __restrict__ ei, float* __restrict__ eex,
                             const float* __restrict__ m, float* __restrict__ den) {
  int idx = blockIdx.x * blockDim.x + threadIdx.x;
  if (idx >= N_EDGES * HEADS) return;
  int e = idx / HEADS, h = idx % HEADS;
  int d = ei[N_EDGES + e];
  float v = __expf(eex[idx] - m[d * HEADS + h]);
  eex[idx] = v;
  atomicAdd(&den[d * HEADS + h], v);
}

// out[dst] += (ex/den[dst]) * xp[src]; one thread per (edge, 4-feature chunk).
__global__ void edge_aggregate(const int* __restrict__ ei, const float* __restrict__ eex,
                               const float* __restrict__ den, const float* __restrict__ xp,
                               float* __restrict__ out) {
  int idx = blockIdx.x * blockDim.x + threadIdx.x;
  if (idx >= N_EDGES * 48) return;
  int e = idx / 48;
  int c = (idx % 48) * 4;
  int h = c >> 6;                          // 64 features per head
  int s = ei[e], d = ei[N_EDGES + e];
  float alpha = eex[e * HEADS + h] / fmaxf(den[d * HEADS + h], 1e-16f);
  float4 xv = *(const float4*)(xp + (long)s * NHID + c);
  float* o = out + (long)d * NHID + c;
  atomicAdd(o + 0, alpha * xv.x);
  atomicAdd(o + 1, alpha * xv.y);
  atomicAdd(o + 2, alpha * xv.z);
  atomicAdd(o + 3, alpha * xv.w);
}

// h = bn(relu(out + bias)) with running stats (eval mode); in-place safe.
__global__ void bias_relu_bn(const float* __restrict__ in_, const float* __restrict__ bias,
                             const float* __restrict__ g, const float* __restrict__ b,
                             const float* __restrict__ mu, const float* __restrict__ var,
                             float* __restrict__ out) {
  int idx = blockIdx.x * blockDim.x + threadIdx.x;
  if (idx >= N_NODES * NHID) return;
  int c = idx % NHID;
  float v = in_[idx] + bias[c];
  v = fmaxf(v, 0.f);
  out[idx] = (v - mu[c]) * (g[c] * rsqrtf(var[c] + BN_EPS)) + b[c];
}

__global__ void pool_accum(const float* __restrict__ h, const int* __restrict__ batch,
                           float* __restrict__ psum, float* __restrict__ pmax) {
  int idx = blockIdx.x * blockDim.x + threadIdx.x;
  if (idx >= N_NODES * NHID) return;
  int n = idx / NHID, c = idx % NHID;
  int g = batch[n];
  float v = h[idx];
  atomicAdd(&psum[g * NHID + c], v);
  atomicMaxF32(&pmax[g * NHID + c], v);
}

// z[g, 0:192] += mean; z[g, 192:384] += max  (accumulates x1+x2+x3)
__global__ void pool_finalize(const float* __restrict__ psum, const float* __restrict__ pmax,
                              const float* __restrict__ cnt, float* __restrict__ z) {
  int idx = blockIdx.x * blockDim.x + threadIdx.x;
  if (idx >= N_GRAPHS * NHID) return;
  int g = idx / NHID, c = idx % NHID;
  z[g * 2 * NHID + c]        += psum[idx] / fmaxf(cnt[g], 1.0f);
  z[g * 2 * NHID + NHID + c] += pmax[idx];
}

// Fused 3-layer MLP head: one block (192 threads) per graph.
__global__ void mlp_head(const float* __restrict__ z,
                         const float* __restrict__ w1, const float* __restrict__ b1,
                         const float* __restrict__ w2, const float* __restrict__ b2,
                         const float* __restrict__ w3, const float* __restrict__ b3,
                         float* __restrict__ out) {
  __shared__ float zin[2 * NHID];
  __shared__ float h1[NHID];
  __shared__ float h2[NHID / 2];
  int g = blockIdx.x, t = threadIdx.x;
  for (int i = t; i < 2 * NHID; i += NHID) zin[i] = z[g * 2 * NHID + i];
  __syncthreads();
  float acc = b1[t];
  for (int i = 0; i < 2 * NHID; ++i) acc += zin[i] * w1[i * NHID + t];
  h1[t] = fmaxf(acc, 0.f);
  __syncthreads();
  if (t < NHID / 2) {
    acc = b2[t];
    for (int i = 0; i < NHID; ++i) acc += h1[i] * w2[i * (NHID / 2) + t];
    h2[t] = fmaxf(acc, 0.f);
  }
  __syncthreads();
  if (t < 2) {
    acc = b3[t];
    for (int i = 0; i < NHID / 2; ++i) acc += h2[i] * w3[i * 2 + t];
    out[g * 2 + t] = acc;
  }
}

static inline int cdiv(long a, int b) { return (int)((a + b - 1) / b); }

extern "C" void kernel_launch(void* const* d_in, const int* in_sizes, int n_in,
                              void* d_out, int out_size, void* d_ws, size_t ws_size,
                              hipStream_t stream) {
  const float* x     = (const float*)d_in[0];
  const int*   ei    = (const int*)d_in[1];
  const int*   batch = (const int*)d_in[3];
  const float* W[3]    = {(const float*)d_in[4],  (const float*)d_in[12], (const float*)d_in[20]};
  const float* asrc[3] = {(const float*)d_in[5],  (const float*)d_in[13], (const float*)d_in[21]};
  const float* adst[3] = {(const float*)d_in[6],  (const float*)d_in[14], (const float*)d_in[22]};
  const float* bias[3] = {(const float*)d_in[7],  (const float*)d_in[15], (const float*)d_in[23]};
  const float* bng[3]  = {(const float*)d_in[8],  (const float*)d_in[16], (const float*)d_in[24]};
  const float* bnb[3]  = {(const float*)d_in[9],  (const float*)d_in[17], (const float*)d_in[25]};
  const float* bnm[3]  = {(const float*)d_in[10], (const float*)d_in[18], (const float*)d_in[26]};
  const float* bnv[3]  = {(const float*)d_in[11], (const float*)d_in[19], (const float*)d_in[27]};
  const float* l1w = (const float*)d_in[28]; const float* l1b = (const float*)d_in[29];
  const float* l2w = (const float*)d_in[30]; const float* l2b = (const float*)d_in[31];
  const float* l3w = (const float*)d_in[32]; const float* l3b = (const float*)d_in[33];
  float* out = (float*)d_out;

  // ---- workspace partition (256B aligned) ----
  char* wsb = (char*)d_ws;
  size_t off = 0;
  auto take = [&](size_t bytes) -> void* {
    off = (off + 255) & ~(size_t)255;
    void* p = wsb + off;
    off += bytes;
    return p;
  };
  float* bufA = (float*)take((size_t)N_NODES * NHID * 4);
  float* bufB = (float*)take((size_t)N_NODES * NHID * 4);   // xp
  float* bufC = (float*)take((size_t)N_NODES * NHID * 4);
  float* als  = (float*)take((size_t)N_NODES * HEADS * 4);
  float* ald  = (float*)take((size_t)N_NODES * HEADS * 4);
  float* mbuf = (float*)take((size_t)N_NODES * HEADS * 4);
  float* den  = (float*)take((size_t)N_NODES * HEADS * 4);
  float* eex  = (float*)take((size_t)N_EDGES * HEADS * 4);
  float* psum = (float*)take((size_t)N_GRAPHS * NHID * 4);
  float* pmax = (float*)take((size_t)N_GRAPHS * NHID * 4);
  float* cnt  = (float*)take((size_t)N_GRAPHS * 4);
  float* zbuf = (float*)take((size_t)N_GRAPHS * 2 * NHID * 4);
  const int wsz0 = NTILES * (N_FEAT / 32) * 32 * 16;   // layer-0 fragments (halves)
  const int wsz12 = NTILES * (NHID / 32) * 32 * 16;
  _Float16* wpk0 = (_Float16*)take((size_t)wsz0 * 2);
  _Float16* wpk1 = (_Float16*)take((size_t)wsz12 * 2);
  _Float16* wpk2 = (_Float16*)take((size_t)wsz12 * 2);

  const int T = 256;
  const float NEG_INF = -__builtin_huge_valf();

  // ---- one-time init ----
  fill_f32<<<cdiv(N_GRAPHS * 2 * NHID, T), T, 0, stream>>>(zbuf, 0.f, N_GRAPHS * 2 * NHID);
  fill_f32<<<1, 64, 0, stream>>>(cnt, 0.f, N_GRAPHS);
  count_nodes<<<cdiv(N_NODES, T), T, 0, stream>>>(batch, cnt);
  repack_w<<<cdiv(wsz0, T),  T, 0, stream>>>(W[0], wpk0, N_FEAT);
  repack_w<<<cdiv(wsz12, T), T, 0, stream>>>(W[1], wpk1, NHID);
  repack_w<<<cdiv(wsz12, T), T, 0, stream>>>(W[2], wpk2, NHID);

  const float* ins[3]   = {x, bufC, bufA};
  float*       outs[3]  = {bufC, bufA, bufC};
  const _Float16* wpk[3] = {wpk0, wpk1, wpk2};
  const int Ks[3] = {N_FEAT, NHID, NHID};

  // one block = 8 M tiles for one 6-N-tile half; x2 halves
  const int gemm_blocks = cdiv(MTILES, 8) * 2;

  for (int l = 0; l < 3; ++l) {
    // xp = in @ W
    gemm_wmma<<<gemm_blocks, T, 0, stream>>>(ins[l], wpk[l], bufB, Ks[l]);
    // attention coefficients
    attn_coeff<<<cdiv(N_NODES * HEADS, T), T, 0, stream>>>(bufB, asrc[l], adst[l], als, ald);
    // per-dst softmax state + aggregation buffer init
    fill_f32<<<cdiv(N_NODES * HEADS, T), T, 0, stream>>>(mbuf, NEG_INF, N_NODES * HEADS);
    fill_f32<<<cdiv(N_NODES * HEADS, T), T, 0, stream>>>(den, 0.f, N_NODES * HEADS);
    fill_f32<<<cdiv((long)N_NODES * NHID, T), T, 0, stream>>>(outs[l], 0.f, N_NODES * NHID);
    // edge passes
    edge_logit_max<<<cdiv(N_EDGES * HEADS, T), T, 0, stream>>>(ei, als, ald, eex, mbuf);
    edge_exp_sum<<<cdiv(N_EDGES * HEADS, T), T, 0, stream>>>(ei, eex, mbuf, den);
    edge_aggregate<<<cdiv((long)N_EDGES * 48, T), T, 0, stream>>>(ei, eex, den, bufB, outs[l]);
    // epilogue: bias + relu + batchnorm (in-place)
    bias_relu_bn<<<cdiv((long)N_NODES * NHID, T), T, 0, stream>>>(
        outs[l], bias[l], bng[l], bnb[l], bnm[l], bnv[l], outs[l]);
    // pooling -> z accumulation
    fill_f32<<<cdiv(N_GRAPHS * NHID, T), T, 0, stream>>>(psum, 0.f, N_GRAPHS * NHID);
    fill_f32<<<cdiv(N_GRAPHS * NHID, T), T, 0, stream>>>(pmax, NEG_INF, N_GRAPHS * NHID);
    pool_accum<<<cdiv((long)N_NODES * NHID, T), T, 0, stream>>>(outs[l], batch, psum, pmax);
    pool_finalize<<<cdiv(N_GRAPHS * NHID, T), T, 0, stream>>>(psum, pmax, cnt, zbuf);
  }

  // classifier head
  mlp_head<<<N_GRAPHS, NHID, 0, stream>>>(zbuf, l1w, l1b, l2w, l2b, l3w, l3b, out);
}
